// RepresentationFusion_12309376270590
// MI455X (gfx1250) — matrix operand order repaired and checked by reading, not compiled, not run on hardware
//
#include <hip/hip_runtime.h>
#include <hip/hip_bf16.h>
#include <math.h>

// ---------------------------------------------------------------------------
// Problem constants (from reference): S=256, B=128, D1=D2=1024, H1=H2=1024
// ---------------------------------------------------------------------------
#define SEQ   256
#define BATCH 128
#define DIM   1024
#define MTOT  (SEQ * BATCH)   // 32768 rows of the big GEMM

typedef __attribute__((ext_vector_type(16))) __bf16 v16bf;
typedef __attribute__((ext_vector_type(8)))  __bf16 v8bf;
typedef __attribute__((ext_vector_type(8)))  float  v8f;

// ---------------- workspace layout (bytes, 256-aligned) --------------------
#define OFF_W1T   0u                       // 1024x1024 bf16, n-major (2 MB)
#define OFF_W2T   2097152u                 // 1024x1024 bf16, n-major (2 MB)
#define OFF_P     4194304u                 // 2 x [128][1024] f32 (1 MB)
#define OFF_S     5242880u                 // 2 x [32768] f32 (256 KB)
#define OFF_STATS 5505024u                 // 2 x 128 float2 (2 KB, pad to 4 KB)
#define OFF_COEF  5509120u                 // 32768 f32 (128 KB)

// Branchless tanh: 1 - 2/(exp(2x)+1); saturates to +-1 for large |x|.
__device__ __forceinline__ float fast_tanh(float x) {
  const float e = __expf(2.0f * x);
  return 1.0f - __fdividef(2.0f, e + 1.0f);
}

// ---------------------------------------------------------------------------
// 1) Convert the "seq" halves of W11/W12 (rows 0..1023) to bf16, transposed
//    to n-major so each WMMA B-fragment is one contiguous 32B run.
// ---------------------------------------------------------------------------
__global__ __launch_bounds__(256) void convert_w_kernel(
    const float* __restrict__ W11, const float* __restrict__ W12,
    __bf16* __restrict__ W1t, __bf16* __restrict__ W2t) {
  const int i = blockIdx.x * 256 + threadIdx.x;   // 0 .. 1M-1
  const int n = i >> 10;
  const int k = i & 1023;
  W1t[i] = (__bf16)W11[k * DIM + n];
  W2t[i] = (__bf16)W12[k * DIM + n];
}

// ---------------------------------------------------------------------------
// 2) Per-batch bias term (the sequence-invariant half of the concat GEMM):
//    P1[b][n] = b11[n] + sum_k feature1[b][k] * W11[(1024+k)*1024 + n]
// ---------------------------------------------------------------------------
__global__ __launch_bounds__(256) void pcalc_kernel(
    const float* __restrict__ f1, const float* __restrict__ f2,
    const float* __restrict__ W11, const float* __restrict__ W12,
    const float* __restrict__ b11, const float* __restrict__ b12,
    float* __restrict__ Pbuf) {
  const int n = blockIdx.x * 256 + threadIdx.x;   // 0..1023
  const int b = blockIdx.y;                       // 0..127
  float a1 = b11[n];
  float a2 = b12[n];
  for (int k = 0; k < DIM; ++k) {
    a1 += f1[b * DIM + k] * W11[(size_t)(DIM + k) * DIM + n];
    a2 += f2[b * DIM + k] * W12[(size_t)(DIM + k) * DIM + n];
  }
  Pbuf[(size_t)b * DIM + n]               = a1;
  Pbuf[(size_t)BATCH * DIM + b * DIM + n] = a2;
}

// ---------------------------------------------------------------------------
// 3) Fused main GEMM (137 GFLOP, bf16 WMMA) + tanh + dot(W2x) + atomic s.
//    BM=128, BN=256, BK=32; 8 waves as 2(wm) x 4(wn); wave tile 64x64 =
//    4x4 WMMA subtiles -> (4+4)*32B/lane per 16 WMMAs = 512 B LDS per WMMA,
//    matching the 64-bank LDS budget (squarer tile than 2x4).
//    LDS double-buffered: A 128x32 bf16 (fp32->bf16 at staging, ds_store),
//    B 256x32 bf16 via global_load_async_to_lds_b128 from pre-transposed
//    bf16 weights (ASYNCcnt + s_wait_asynccnt + split barrier).
//    Epilogue: h = tanh(acc + P[b][n]); s[mat][row] += h * W2x[n]
//    (b21/b22 omitted: constant shifts cancel in the sequence softmax.)
// ---------------------------------------------------------------------------
__global__ __launch_bounds__(256) void gemm_tanh_dot_kernel(
    const float*  __restrict__ Aseq,   // [32768][1024] fp32
    const __bf16* __restrict__ W1t,    // [1024 n][1024 k] bf16
    const __bf16* __restrict__ W2t,
    const float*  __restrict__ Pbuf,   // [2][128][1024] fp32
    const float*  __restrict__ W21,    // [1024] fp32
    const float*  __restrict__ W22,    // [1024] fp32
    float* __restrict__ sbuf) {        // [2][32768] fp32 (pre-zeroed)
  __shared__ __bf16 As[2][128 * 32];   // 2 x 8 KB
  __shared__ __bf16 Bs[2][256 * 32];   // 2 x 16 KB

  const int t    = threadIdx.x;
  const int lane = t & 31;
  const int wave = t >> 5;
  const int wm   = wave >> 2;          // 0..1  (64 rows each)
  const int wn   = wave & 3;           // 0..3  (64 cols each)
  const int half = lane >> 4;          // ISA A/B/C lane-half
  const int l15  = lane & 15;

  const int tileN   = blockIdx.x;               // 0..7 over Ntot=2048
  const int tileM   = blockIdx.y;               // 0..255 over M=32768
  const int mat     = (tileN >= 4) ? 1 : 0;
  const int NblkLoc = (tileN & 3) * 256;        // n base within matrix
  const int Mblk    = tileM * 128;

  const __bf16* __restrict__ Wt = mat ? W2t : W1t;
  const float*  __restrict__ P  = Pbuf + (size_t)mat * BATCH * DIM;
  const float*  __restrict__ wv = mat ? W22 : W21;
  float* sdst                   = sbuf + (size_t)mat * MTOT;

  // ---- tile staging: A (load fp32, cvt, ds_store) + B (async -> LDS)
  auto stage = [&](int sb, int kk) {
    {
      const int r  = t >> 1;             // 0..127
      const int kc = (t & 1) * 16;       // 0 or 16
      const float* gp = Aseq + (size_t)(Mblk + r) * DIM + kk + kc;
      const float4 f0 = *(const float4*)(gp);
      const float4 f1 = *(const float4*)(gp + 4);
      const float4 f2 = *(const float4*)(gp + 8);
      const float4 f3 = *(const float4*)(gp + 12);
      v16bf v;
      v[0]  = (__bf16)f0.x; v[1]  = (__bf16)f0.y;
      v[2]  = (__bf16)f0.z; v[3]  = (__bf16)f0.w;
      v[4]  = (__bf16)f1.x; v[5]  = (__bf16)f1.y;
      v[6]  = (__bf16)f1.z; v[7]  = (__bf16)f1.w;
      v[8]  = (__bf16)f2.x; v[9]  = (__bf16)f2.y;
      v[10] = (__bf16)f2.z; v[11] = (__bf16)f2.w;
      v[12] = (__bf16)f3.x; v[13] = (__bf16)f3.y;
      v[14] = (__bf16)f3.z; v[15] = (__bf16)f3.w;
      *(v16bf*)&As[sb][r * 32 + kc] = v;
    }
    // B: one 64B row (n = t) per thread as 4 async 16B transfers -> 16 KB.
    {
      const char* gb = (const char*)(Wt + (size_t)(NblkLoc + t) * DIM + kk);
      const unsigned la = (unsigned)(size_t)(&Bs[sb][t * 32]);
#pragma unroll
      for (int q = 0; q < 4; ++q) {
        asm volatile("global_load_async_to_lds_b128 %0, %1, off"
                     :: "v"(la + q * 16), "v"(gb + q * 16) : "memory");
      }
    }
  };

  v8f acc[4][4];
#pragma unroll
  for (int i = 0; i < 4; ++i)
#pragma unroll
    for (int j = 0; j < 4; ++j) acc[i][j] = v8f{};

  stage(0, 0);

  for (int k0 = 0; k0 < DIM; k0 += 32) {
    const int buf = (k0 >> 5) & 1;
    // my async B-stage done; barrier: everyone's A+B staging visible, and
    // it also fences last iteration's readers from this iteration's stage.
    asm volatile("s_wait_asynccnt 0x0" ::: "memory");
    __syncthreads();
    if (k0 + 32 < DIM) stage(buf ^ 1, k0 + 32);

    // ---- A fragments (16-bit A 16x32): half h -> K = h*16..h*16+15
    v16bf afrag[4];
#pragma unroll
    for (int ms = 0; ms < 4; ++ms) {
      const int row = wm * 64 + ms * 16 + l15;
      afrag[ms] = *(const v16bf*)&As[buf][row * 32 + half * 16];
    }
    // ---- B fragments: lanes 0-15 N=lane K=0..15; lanes 16-31 K=16..31.
    // Interleave each B fragment with its 4 WMMAs to bound register pressure.
#pragma unroll
    for (int ns = 0; ns < 4; ++ns) {
      const int n = wn * 64 + ns * 16 + l15;
      const v16bf bfrag = *(const v16bf*)&Bs[buf][n * 32 + half * 16];
#pragma unroll
      for (int ms = 0; ms < 4; ++ms)
        acc[ms][ns] = __builtin_amdgcn_wmma_f32_16x16x32_bf16(
            false, afrag[ms], false, bfrag, (short)0, acc[ms][ns],
            false, false);
    }
  }

  // ---- Epilogue: C/D layout -> VGPR v holds row (half*8+v), col l15.
#pragma unroll
  for (int ms = 0; ms < 4; ++ms) {
#pragma unroll
    for (int ns = 0; ns < 4; ++ns) {
      const int n = NblkLoc + wn * 64 + ns * 16 + l15;
      const float wcol = wv[n];
#pragma unroll
      for (int v = 0; v < 8; ++v) {
        const int row  = Mblk + wm * 64 + ms * 16 + half * 8 + v;
        const int bidx = row & (BATCH - 1);
        const float h = fast_tanh(acc[ms][ns][v] + P[(size_t)bidx * DIM + n]);
        float c = h * wcol;
        // reduce over the 16 N-lanes of this half-wave (same row)
        c += __shfl_xor(c, 8, 16);
        c += __shfl_xor(c, 4, 16);
        c += __shfl_xor(c, 2, 16);
        c += __shfl_xor(c, 1, 16);
        if (l15 == 0) atomicAdd(&sdst[row], c);
      }
    }
  }
}

// ---------------------------------------------------------------------------
// 4) Softmax stats over the sequence axis: per (mat, b): max and 1/sumexp.
// ---------------------------------------------------------------------------
__global__ void stats_kernel(const float* __restrict__ sbuf,
                             float2* __restrict__ stats) {
  const int t   = threadIdx.x;        // 256 threads, 1 block
  const int mat = t >> 7;
  const int b   = t & (BATCH - 1);
  const float* sp = sbuf + (size_t)mat * MTOT;
  float m = -INFINITY;
  for (int s = 0; s < SEQ; ++s) m = fmaxf(m, sp[s * BATCH + b]);
  float l = 0.0f;
  for (int s = 0; s < SEQ; ++s) l += __expf(sp[s * BATCH + b] - m);
  stats[t] = make_float2(m, 1.0f / l);
}

// ---------------------------------------------------------------------------
// 5) Combined weight per (s,b):  coef = (softmax1 + softmax2) * 0.5 / S
// ---------------------------------------------------------------------------
__global__ __launch_bounds__(256) void coef_kernel(
    const float* __restrict__ sbuf, const float2* __restrict__ stats,
    float* __restrict__ coef) {
  const int idx = blockIdx.x * 256 + threadIdx.x;  // 0..32767 (= s*128+b)
  const int b   = idx & (BATCH - 1);
  const float2 st1 = stats[b];
  const float2 st2 = stats[BATCH + b];
  const float w1 = __expf(sbuf[idx] - st1.x) * st1.y;
  const float w2 = __expf(sbuf[MTOT + idx] - st2.x) * st2.y;
  coef[idx] = (w1 + w2) * (0.5f / (float)SEQ);
}

// ---------------------------------------------------------------------------
// 6) Output: out[b][d] = sum_s coef[s,b] * seq[s,b,d]   (streams 128 MB once)
// ---------------------------------------------------------------------------
__global__ __launch_bounds__(256) void out_kernel(
    const float* __restrict__ seq, const float* __restrict__ coef,
    float* __restrict__ out) {
  const int d = blockIdx.x * 256 + threadIdx.x;    // 0..1023
  const int b = blockIdx.y;                        // 0..127
  float acc = 0.0f;
  for (int s = 0; s < SEQ; ++s)
    acc += coef[s * BATCH + b] * seq[(size_t)(s * BATCH + b) * DIM + d];
  out[(size_t)b * DIM + d] = acc;
}

// ---------------------------------------------------------------------------
extern "C" void kernel_launch(void* const* d_in, const int* in_sizes, int n_in,
                              void* d_out, int out_size, void* d_ws,
                              size_t ws_size, hipStream_t stream) {
  const float* f1   = (const float*)d_in[0];   // feature1   [128,1024]
  const float* f2   = (const float*)d_in[1];   // feature2   [128,1024]
  const float* seq  = (const float*)d_in[2];   // feature1_seq [256,128,1024]
  const float* W11  = (const float*)d_in[3];   // [2048,1024]
  const float* b11  = (const float*)d_in[4];   // [1024]
  const float* W12  = (const float*)d_in[5];   // [2048,1024]
  const float* b12  = (const float*)d_in[6];   // [1024]
  const float* W21  = (const float*)d_in[7];   // [1024,1]
  // d_in[8] = b21, d_in[10] = b22: constant shifts, cancel in softmax.
  const float* W22  = (const float*)d_in[9];   // [1024,1]

  char* ws = (char*)d_ws;
  __bf16* W1t   = (__bf16*)(ws + OFF_W1T);
  __bf16* W2t   = (__bf16*)(ws + OFF_W2T);
  float*  Pbuf  = (float*)(ws + OFF_P);
  float*  sbuf  = (float*)(ws + OFF_S);
  float2* stats = (float2*)(ws + OFF_STATS);
  float*  coef  = (float*)(ws + OFF_COEF);

  // zero the atomic s accumulators (capturable stream op)
  hipMemsetAsync(sbuf, 0, 2 * (size_t)MTOT * sizeof(float), stream);

  convert_w_kernel<<<(DIM * DIM) / 256, 256, 0, stream>>>(W11, W12, W1t, W2t);
  pcalc_kernel<<<dim3(DIM / 256, BATCH), 256, 0, stream>>>(f1, f2, W11, W12,
                                                           b11, b12, Pbuf);
  gemm_tanh_dot_kernel<<<dim3(8, MTOT / 128), 256, 0, stream>>>(
      seq, W1t, W2t, Pbuf, W21, W22, sbuf);
  stats_kernel<<<1, 256, 0, stream>>>(sbuf, stats);
  coef_kernel<<<MTOT / 256, 256, 0, stream>>>(sbuf, stats, coef);
  out_kernel<<<dim3(DIM / 256, BATCH), 256, 0, stream>>>(seq, coef,
                                                         (float*)d_out);
}
